// CompletionNet_energy2_5_27006754357607
// MI455X (gfx1250) — matrix operand order repaired and checked by reading, not compiled, not run on hardware
//
#include <hip/hip_runtime.h>
#include <hip/hip_bf16.h>

// ---------------------------------------------------------------------------
// CDNA5 / gfx1250 CompletionNet forward.
// Convs/tconvs = implicit GEMM on v_wmma_f32_16x16x32_bf16 (wave32).
// All activation buffers use a channel stride Cp = max(32, C) padded with
// zeros, and padding taps read from a zero page -> the inner loop is
// branch-free: 4x global_load_b128 (A) + 2x ds_load_b128 (B) + 2x v_wmma.
// B slabs are staged Global->LDS by the Tensor Data Mover (tensor_load_to_lds
// + s_wait_tensorcnt), keeping compute waves' issue slots free for WMMA.
// ---------------------------------------------------------------------------

typedef __attribute__((ext_vector_type(16))) __bf16 v16bf;
typedef __attribute__((ext_vector_type(8)))  __bf16 v8bf;
typedef __attribute__((ext_vector_type(8)))  float  v8f;
typedef __attribute__((ext_vector_type(4)))  unsigned int u32x4;
typedef __attribute__((ext_vector_type(4)))  int          i32x4;
typedef __attribute__((ext_vector_type(8)))  int          i32x8;

// ---------------------------------------------------------------------------
// TDM: DMA a contiguous run of `nelem` bf16 from global to LDS.
// D# layout per CDNA5 ISA (1-row 2D tile).
// ---------------------------------------------------------------------------
__device__ __forceinline__
void tdm_load_slab(const void* gsrc, void* lds_dst, unsigned nelem)
{
    const unsigned long long ga = (unsigned long long)(uintptr_t)gsrc;
    const unsigned lds = (unsigned)(uintptr_t)lds_dst;  // addr[31:0] = LDS offset

    u32x4 g0;
    g0[0] = 1u;                                   // count=1, user descriptor
    g0[1] = lds;                                  // lds_addr (bytes)
    g0[2] = (unsigned)ga;                         // global_addr[31:0]
    g0[3] = ((unsigned)(ga >> 32) & 0x01FFFFFFu)  // global_addr[56:32]
            | (2u << 30);                         // type = 2 ("image")

    i32x8 g1 = {};
    g1[0] = (int)(1u << 16);                        // data_size=1 (2 bytes); wg_mask=0
    g1[1] = (int)((nelem & 0xFFFFu) << 16);         // tensor_dim0[15:0]  (bits 63:48)
    g1[2] = (int)(((nelem >> 16) & 0xFFFFu)         // tensor_dim0[31:16] (bits 79:64)
                  | (1u << 16));                    // tensor_dim1 = 1    (bits 95:80)
    g1[3] = (int)((nelem & 0xFFFFu) << 16);         // tile_dim0          (bits 127:112)
    g1[4] = 1;                                      // tile_dim1 = 1      (bits 143:128)
    g1[5] = (int)nelem;                             // tensor_dim0_stride (bits 191:160)

    i32x4 gz = {};
#if __clang_major__ >= 23
    i32x8 gz8 = {};
    __builtin_amdgcn_tensor_load_to_lds(g0, g1, gz, gz, gz8, 0);
#else
    __builtin_amdgcn_tensor_load_to_lds(g0, g1, gz, gz, 0);
#endif
}

// ---------------------------------------------------------------------------
// Weight transpose + convert + K-pad:
//   w[tap][Ci][Co] (f32)  ->  o[tap][Co][Cp] (bf16), zero for k >= Ci
// ---------------------------------------------------------------------------
__global__ __launch_bounds__(256)
void wt_transpose_bf16(const float* __restrict__ w, __bf16* __restrict__ o,
                       int taps, int Ci, int Co, int Cp)
{
    const long long i = (long long)blockIdx.x * 256 + threadIdx.x;
    const long long total = (long long)taps * Co * Cp;
    if (i >= total) return;
    const int k   = (int)(i % Cp);
    const long long t = i / Cp;
    const int n   = (int)(t % Co);
    const int tap = (int)(t / Co);
    o[i] = (k < Ci) ? (__bf16)w[((long long)tap * Ci + k) * Co + n] : (__bf16)0.0f;
}

__global__ __launch_bounds__(256)
void zero_bf16(__bf16* __restrict__ p, int n)
{
    const int i = blockIdx.x * 256 + threadIdx.x;
    if (i < n) p[i] = (__bf16)0.0f;
}

// ---------------------------------------------------------------------------
// Implicit-GEMM 3D conv / transpose-conv.
//   in : bf16 NDHWC activations, channel stride Cp (mult of 32, zero-padded)
//   wt2: bf16 [tap][Co][Cp] transposed weights (K zero-padded)
//   zp : zero page (>= Cp bf16 zeros) for padding taps
//   out: f32 NDHWC, channel stride Co (raw pre-BN conv result)
// Block = 256 threads = 8 waves; each wave owns TWO 16x16 tiles (32 M-rows),
// all waves share one 16-column N-tile; the tap's B slab (16 x Cp bf16) is
// DMA'd into LDS by the TDM (wave 0), published with s_wait_tensorcnt+barrier.
// A-frag (16-bit 16x32): lane L holds M=L&15, K in {g..g+7, 16+g..16+g+7},
//   g=(L>=16)?8:0.  B-frag (32x16): lane L holds N=L&15, same K as A-frag.
// D: VGPR r holds M=(L>=16?8:0)+r, N=L&15.
// ---------------------------------------------------------------------------
__global__ __launch_bounds__(256)
void conv3d_wmma_bf16(const __bf16* __restrict__ in, const __bf16* __restrict__ wt2,
                      const __bf16* __restrict__ zp, float* __restrict__ out,
                      int B, int Di, int Hi, int Wi, int Cp,
                      int Do, int Ho, int Wo, int Co,
                      int KD, int KH, int KW, int stride, int pad, int transposed)
{
    __shared__ __bf16 lB[16 * 1024];          // 32 KB: one tap's 16 x Cp slab

    const int tid  = threadIdx.x;
    const int lane = tid & 31;
    const int wave = tid >> 5;
    const int Mtotal = B * Do * Ho * Wo;
    const int mBase = (blockIdx.x * 8 + wave) * 32;   // 2 tiles per wave
    const int nBase = blockIdx.y * 16;

    const int g   = (lane >> 4) * 8;          // A K-group offset within 32-chunk
    const int kbB = (lane >> 4) * 16;         // unused: B loads use g, like A

    // ---- decode this lane's two A-rows (output sites) ----
    int xo[2], yo[2], zo[2], bb[2];
    bool vm[2];
#pragma unroll
    for (int t = 0; t < 2; ++t) {
        const int m = mBase + t * 16 + (lane & 15);
        vm[t] = (m < Mtotal);
        const int mm = vm[t] ? m : 0;
        xo[t] = mm % Wo;  int t1 = mm / Wo;
        yo[t] = t1 % Ho;  int t2 = t1 / Ho;
        zo[t] = t2 % Do;  bb[t] = t2 / Do;
    }

    v8f acc0 = {}, acc1 = {};
    const int taps = KD * KH * KW;
    const unsigned slab = (unsigned)(16 * Cp);        // bf16 elements per tap slab
    const __bf16* lrow = lB + (lane & 15) * Cp;

    for (int tap = 0; tap < taps; ++tap) {
        // ---- TDM: DMA this tap's B slab wt2[tap][nBase..+16][*] into LDS ----
        if (wave == 0) {
            tdm_load_slab(wt2 + ((long long)tap * Co + nBase) * Cp, lB, slab);
            __builtin_amdgcn_s_wait_tensorcnt(0);
        }
        __syncthreads();

        // ---- map output sites -> input sites for this tap ----
        const int kw = tap % KW;  int tt = tap / KW;
        const int kh = tt % KH;   const int kd = tt / KH;

        const __bf16* ap[2];
#pragma unroll
        for (int t = 0; t < 2; ++t) {
            int zi, yi, xi;
            bool ok = vm[t];
            if (!transposed) {
                zi = zo[t] * stride - pad + kd;
                yi = yo[t] * stride - pad + kh;
                xi = xo[t] * stride - pad + kw;
                ok = ok && (zi >= 0) && (zi < Di) && (yi >= 0) && (yi < Hi)
                        && (xi >= 0) && (xi < Wi);
            } else {  // stride-2 transpose conv: gather from contributing parity
                const int zn = zo[t] + pad - kd, yn = yo[t] + pad - kh,
                          xn = xo[t] + pad - kw;
                ok = ok && (zn >= 0) && !(zn & 1) && (yn >= 0) && !(yn & 1)
                        && (xn >= 0) && !(xn & 1);
                zi = zn >> 1; yi = yn >> 1; xi = xn >> 1;
                ok = ok && (zi < Di) && (yi < Hi) && (xi < Wi);
            }
            // zero-page instead of predicated loads: loop stays branch-free
            ap[t] = ok
                ? in + ((((long long)bb[t] * Di + zi) * Hi + yi) * (long long)Wi + xi) * Cp
                : zp;
        }

        for (int c0 = 0; c0 < Cp; c0 += 32) {
            // ---- A fragments: unconditional 128-bit global loads ----
            v8bf t00 = *(const v8bf*)(ap[0] + c0 + g);
            v8bf t01 = *(const v8bf*)(ap[0] + c0 + g + 16);
            v8bf t10 = *(const v8bf*)(ap[1] + c0 + g);
            v8bf t11 = *(const v8bf*)(ap[1] + c0 + g + 16);
            // ---- B fragment: two 128-bit LDS loads (ds_load_b128) ----
            v8bf b0 = *(const v8bf*)(lrow + c0 + g);
            v8bf b1 = *(const v8bf*)(lrow + c0 + g + 16);

            v16bf a0, a1, bf;
#pragma unroll
            for (int e = 0; e < 8; ++e) {
                a0[e] = t00[e]; a0[8 + e] = t01[e];
                a1[e] = t10[e]; a1[8 + e] = t11[e];
                bf[e] = b0[e];  bf[8 + e] = b1[e];
            }
            acc0 = __builtin_amdgcn_wmma_f32_16x16x32_bf16(
                       false, a0, false, bf, (short)0, acc0, false, false);
            acc1 = __builtin_amdgcn_wmma_f32_16x16x32_bf16(
                       false, a1, false, bf, (short)0, acc1, false, false);
        }
        __syncthreads();   // slab reused next tap
    }

    // ---- store D tiles ----
    const int nOut = nBase + (lane & 15);
    const int mrb  = (lane >> 4) * 8;
#pragma unroll
    for (int r = 0; r < 8; ++r) {
        const int mo0 = mBase + mrb + r;
        if (mo0 < Mtotal) out[(long long)mo0 * Co + nOut] = acc0[r];
        const int mo1 = mBase + 16 + mrb + r;
        if (mo1 < Mtotal) out[(long long)mo1 * Co + nOut] = acc1[r];
    }
}

// ---------------------------------------------------------------------------
// Per-channel mean/var over all sites (train-mode BN). One block per channel.
// ---------------------------------------------------------------------------
__global__ __launch_bounds__(256)
void bn_stats(const float* __restrict__ x, float* __restrict__ stats,
              int sites, int C)
{
    const int c = blockIdx.x;
    float s = 0.f, s2 = 0.f;
    for (int i = threadIdx.x; i < sites; i += 256) {
        const float v = x[(long long)i * C + c];
        s += v; s2 += v * v;
    }
    __shared__ float sh[256], sh2[256];
    sh[threadIdx.x] = s; sh2[threadIdx.x] = s2;
    __syncthreads();
    for (int off = 128; off > 0; off >>= 1) {
        if (threadIdx.x < off) {
            sh[threadIdx.x]  += sh[threadIdx.x + off];
            sh2[threadIdx.x] += sh2[threadIdx.x + off];
        }
        __syncthreads();
    }
    if (threadIdx.x == 0) {
        const float inv = 1.0f / (float)sites;
        const float mean = sh[0] * inv;
        const float var  = sh2[0] * inv - mean * mean;
        stats[c * 2 + 0] = mean;
        stats[c * 2 + 1] = var;
    }
}

// ---------------------------------------------------------------------------
// Fused BN + ReLU (+ optional skip-add): f32 (stride Co) -> bf16 (stride Cp,
// pad channels zeroed so downstream convs read zeros).
// ---------------------------------------------------------------------------
__global__ __launch_bounds__(256)
void bn_relu_apply(const float* __restrict__ x, const float* __restrict__ gamma,
                   const float* __restrict__ beta, const float* __restrict__ stats,
                   const __bf16* __restrict__ skip, __bf16* __restrict__ y,
                   long long sites, int Co, int Cp)
{
    const long long i = (long long)blockIdx.x * blockDim.x + threadIdx.x;
    if (i >= sites * Cp) return;
    const int c = (int)(i % Cp);
    const long long s = i / Cp;
    float v = 0.0f;
    if (c < Co) {
        const float mean = stats[c * 2 + 0];
        const float var  = stats[c * 2 + 1];
        v = gamma[c] * (x[s * Co + c] - mean) * rsqrtf(var + 1e-5f) + beta[c];
        v = fmaxf(v, 0.0f);
        if (skip) v += (float)skip[i];
    }
    y[i] = (__bf16)v;
}

// ---------------------------------------------------------------------------
// Per-scale cls head + MinkowskiPruning: zero sites where logit1 <= logit0.
// ---------------------------------------------------------------------------
__global__ __launch_bounds__(256)
void head_prune(__bf16* __restrict__ d, const float* __restrict__ w,
                const float* __restrict__ bias, int sites, int Co, int Cp)
{
    const int s = blockIdx.x * blockDim.x + threadIdx.x;
    if (s >= sites) return;
    __bf16* p = d + (long long)s * Cp;
    float a0 = bias[0], a1 = bias[1];
    for (int c = 0; c < Co; ++c) {
        const float v = (float)p[c];
        a0 += v * w[c * 2 + 0];
        a1 += v * w[c * 2 + 1];
    }
    if (!(a1 > a0))
        for (int c = 0; c < Co; ++c) p[c] = (__bf16)0.0f;
}

// ---------------------------------------------------------------------------
// Final heads: sigmoid(cls0), sigmoid(cls1), relu(energy) -> f32 [..,3]
// ---------------------------------------------------------------------------
__global__ __launch_bounds__(256)
void final_heads(const __bf16* __restrict__ d,
                 const float* __restrict__ wc, const float* __restrict__ bc,
                 const float* __restrict__ we, const float* __restrict__ be,
                 float* __restrict__ out, int sites, int Co, int Cp)
{
    const int s = blockIdx.x * blockDim.x + threadIdx.x;
    if (s >= sites) return;
    const __bf16* p = d + (long long)s * Cp;
    float c0 = bc[0], c1 = bc[1], en = be[0];
    for (int c = 0; c < Co; ++c) {
        const float v = (float)p[c];
        c0 += v * wc[c * 2 + 0];
        c1 += v * wc[c * 2 + 1];
        en += v * we[c];
    }
    out[(long long)s * 3 + 0] = 1.0f / (1.0f + expf(-c0));
    out[(long long)s * 3 + 1] = 1.0f / (1.0f + expf(-c1));
    out[(long long)s * 3 + 2] = fmaxf(en, 0.0f);
}

// x [sites] f32 -> padded bf16 [sites][32], channel 0 = x, rest 0
__global__ __launch_bounds__(256)
void cvt_pad_x(const float* __restrict__ x, __bf16* __restrict__ y, long long sites)
{
    const long long i = (long long)blockIdx.x * blockDim.x + threadIdx.x;
    if (i >= sites * 32) return;
    const int c = (int)(i & 31);
    y[i] = (c == 0) ? (__bf16)x[i >> 5] : (__bf16)0.0f;
}

// ---------------------------------------------------------------------------
// Host orchestration
// ---------------------------------------------------------------------------
extern "C" void kernel_launch(void* const* d_in, const int* in_sizes, int n_in,
                              void* d_out, int out_size, void* d_ws, size_t ws_size,
                              hipStream_t stream)
{
    (void)in_sizes; (void)n_in; (void)out_size; (void)ws_size;

    static const int ENCC[7] = {16, 32, 64, 128, 256, 512, 1024};
    static const int DS_s[6]  = {5, 4, 3, 2, 1, 0};
    static const int DS_ci[6] = {1024, 512, 256, 128, 64, 32};
    static const int DS_co[6] = {512, 256, 128, 64, 32, 16};
    static const int DS_k[6]  = {4, 2, 2, 2, 2, 2};

    auto W  = [&](int i) { return (const float*)d_in[i]; };
    auto cp = [](int c)  { return c < 32 ? 32 : c; };
    const float* X = (const float*)d_in[0];

    // ---- workspace layout (256B aligned) ----
    char* ws = (char*)d_ws;
    size_t off = 0;
    auto balloc = [&](size_t bytes) -> char* {
        char* p = ws + off;
        off += (bytes + 255) & ~(size_t)255;
        return p;
    };
    long long sites[7];
    for (int i = 0; i < 7; ++i) {
        const long long sp = 64 >> i;
        sites[i] = 2LL * sp * sp * sp;
    }
    __bf16* skipb[7];
    for (int i = 0; i < 7; ++i)
        skipb[i] = (__bf16*)balloc((size_t)(sites[i] * cp(ENCC[i])) * sizeof(__bf16));
    float*  scratch = (float*) balloc((size_t)8388608  * sizeof(float));    // max sites*Co
    __bf16* bufA    = (__bf16*)balloc((size_t)16777216 * sizeof(__bf16));   // max sites*Cp
    __bf16* bufB    = (__bf16*)balloc((size_t)16777216 * sizeof(__bf16));
    __bf16* wtmp    = (__bf16*)balloc((size_t)33554432 * sizeof(__bf16));   // max taps*Co*Cp
    __bf16* zp      = (__bf16*)balloc((size_t)1024     * sizeof(__bf16));   // zero page
    float*  stats   = (float*) balloc((size_t)2048     * sizeof(float));    // [C][mean,var]

    zero_bf16<<<dim3(4), 256, 0, stream>>>(zp, 1024);

    auto run_conv = [&](const __bf16* in, const float* w, float* outp,
                        int Bn, int spi, int Ci, int spo, int Co,
                        int K, int stride, int pad, int transposed) {
        const int Cp = cp(Ci);
        const int taps = K * K * K;
        const long long wtot = (long long)taps * Co * Cp;
        wt_transpose_bf16<<<dim3((unsigned)((wtot + 255) / 256)), 256, 0, stream>>>(
            w, wtmp, taps, Ci, Co, Cp);
        const int Mtotal = Bn * spo * spo * spo;
        dim3 grid((Mtotal + 255) / 256, (Co + 15) / 16);
        conv3d_wmma_bf16<<<grid, 256, 0, stream>>>(
            in, wtmp, zp, outp, Bn, spi, spi, spi, Cp, spo, spo, spo, Co,
            K, K, K, stride, pad, transposed);
    };
    auto run_bn = [&](const float* scr, const float* g, const float* b,
                      const __bf16* skip, __bf16* outp, long long nSites, int Co) {
        bn_stats<<<dim3(Co), 256, 0, stream>>>(scr, stats, (int)nSites, Co);
        const long long total = nSites * cp(Co);
        bn_relu_apply<<<dim3((unsigned)((total + 255) / 256)), 256, 0, stream>>>(
            scr, g, b, stats, skip, outp, nSites, Co, cp(Co));
    };

    // ---- input f32 -> padded bf16 (bufB is free at this point) ----
    cvt_pad_x<<<dim3((unsigned)((sites[0] * 32 + 255) / 256)), 256, 0, stream>>>(
        X, bufB, sites[0]);

    // ---- encoder stage s1: conv3 (1->16) + BN + ReLU ----
    run_conv(bufB, W(1), scratch, 2, 64, 1, 64, 16, 3, 1, 1, 0);
    run_bn(scratch, W(2), W(3), nullptr, skipb[0], sites[0], 16);

    // ---- encoder stages i=1..6: k2s2 down + k3s1, each BN+ReLU ----
    int sp = 64;
    for (int i = 1; i <= 6; ++i) {
        const int ci = ENCC[i - 1], co = ENCC[i];
        const int base = 4 + (i - 1) * 6;
        const int spo = sp >> 1;
        run_conv(skipb[i - 1], W(base + 0), scratch, 2, sp, ci, spo, co, 2, 2, 0, 0);
        run_bn(scratch, W(base + 1), W(base + 2), nullptr, bufA, sites[i], co);
        run_conv(bufA, W(base + 3), scratch, 2, spo, co, spo, co, 3, 1, 1, 0);
        run_bn(scratch, W(base + 4), W(base + 5), nullptr, skipb[i], sites[i], co);
        sp = spo;
    }

    // ---- decoder: tconv + conv3 + skip-add + cls-head pruning per scale ----
    const __bf16* d = skipb[6];
    for (int j = 0; j < 6; ++j) {
        const int s = DS_s[j], ci = DS_ci[j], co = DS_co[j], k = DS_k[j];
        const int base = 40 + j * 10;
        const int spi = 64 >> (s + 1), spo = 64 >> s;
        const long long nSites = sites[s];
        const int padT = (k == 4) ? 1 : 0;
        run_conv(d, W(base + 0), scratch, 2, spi, ci, spo, co, k, 2, padT, 1);
        run_bn(scratch, W(base + 1), W(base + 2), nullptr, bufA, nSites, co);
        run_conv(bufA, W(base + 3), scratch, 2, spo, co, spo, co, 3, 1, 1, 0);
        run_bn(scratch, W(base + 4), W(base + 5), skipb[s], bufB, nSites, co);
        head_prune<<<dim3((unsigned)((nSites + 255) / 256)), 256, 0, stream>>>(
            bufB, W(base + 6), W(base + 7), (int)nSites, co, cp(co));
        d = bufB;
    }

    // ---- final heads at s1 ----
    final_heads<<<dim3((unsigned)((sites[0] + 255) / 256)), 256, 0, stream>>>(
        bufB, W(100), W(101), W(102), W(103), (float*)d_out, (int)sites[0], 16, 32);
}
